// MultiEncoderTopKSAE_16939351015445
// MI455X (gfx1250) — matrix operand order repaired and checked by reading, not compiled, not run on hardware
//
#include <hip/hip_runtime.h>
#include <hip/hip_bf16.h>

#define D_MODEL   1024
#define N_GROUPS  8
#define SUB_SIZE  8192
#define TOP_K     32
#define BATCH     2048

#define TILE_M    16                      // batch rows per block
#define TILE_N    128                     // encoder columns per chunk
#define KCHUNK    32                      // bf16 WMMA K
#define NS_CHUNKS (SUB_SIZE / TILE_N)     // 64
#define NK_CHUNKS (D_MODEL / KCHUNK)      // 32
#define D8        (D_MODEL / 8)           // uint4s per row of bf16 data

typedef __attribute__((ext_vector_type(16))) __bf16 v16bf;
typedef __attribute__((ext_vector_type(8)))  float  v8f;

union BFrag { v16bf v; uint4 q[2]; };

// fp32 -> (bf16 hi, bf16 lo) split via truncation; |lo| < 2^-7 |x|
__device__ __forceinline__ void split_bf16(float f, unsigned short& h, unsigned short& l) {
    unsigned u = __float_as_uint(f);
    h = (unsigned short)(u >> 16);
    float fh = __uint_as_float((unsigned)h << 16);
    float r  = f - fh;
    l = (unsigned short)(__float_as_uint(r) >> 16);
}

// =====================================================================
// Pass 0: W_enc [g][d][s] fp32  ->  Wt_hi/Wt_lo [g][s][d] bf16 (transposed)
// =====================================================================
__global__ __launch_bounds__(256) void convert_wenc(
    const float* __restrict__ W_enc,
    unsigned short* __restrict__ WtH,
    unsigned short* __restrict__ WtL)
{
    __shared__ float tile[64][65];                 // padded to dodge bank conflicts
    const int bid   = blockIdx.x;
    const int g     = bid >> 11;                   // 16*128 = 2048 tiles per group
    const int rem   = bid & 2047;
    const int d0    = (rem >> 7) << 6;             // 16 d-tiles of 64
    const int s0    = (rem & 127) << 6;            // 128 s-tiles of 64
    const int tid   = threadIdx.x;

    // load 64x64 fp32 tile (rows = d, coalesced in s)
    {
        const int dd = tid >> 2;
        const int c0 = (tid & 3) * 16;
        const float* src = W_enc + (size_t)g * D_MODEL * SUB_SIZE
                         + (size_t)(d0 + dd) * SUB_SIZE + s0 + c0;
        #pragma unroll
        for (int j = 0; j < 16; ++j) tile[dd][c0 + j] = src[j];
    }
    __syncthreads();

    // write transposed: rows = s, 16 d-values per thread, packed bf16 hi/lo
    {
        const int ss = tid >> 2;
        const int jb = (tid & 3) * 16;
        unsigned hw[8], lw[8];
        #pragma unroll
        for (int p = 0; p < 8; ++p) {
            unsigned short h0, l0, h1, l1;
            split_bf16(tile[jb + 2*p    ][ss], h0, l0);
            split_bf16(tile[jb + 2*p + 1][ss], h1, l1);
            hw[p] = (unsigned)h0 | ((unsigned)h1 << 16);
            lw[p] = (unsigned)l0 | ((unsigned)l1 << 16);
        }
        const size_t off = (size_t)g * SUB_SIZE * D_MODEL
                         + (size_t)(s0 + ss) * D_MODEL + d0 + jb;   // shorts, 32B aligned
        uint4* dsth = (uint4*)(WtH + off);
        uint4* dstl = (uint4*)(WtL + off);
        dsth[0] = make_uint4(hw[0], hw[1], hw[2], hw[3]);
        dsth[1] = make_uint4(hw[4], hw[5], hw[6], hw[7]);
        dstl[0] = make_uint4(lw[0], lw[1], lw[2], lw[3]);
        dstl[1] = make_uint4(lw[4], lw[5], lw[6], lw[7]);
    }
}

// =====================================================================
// Fused encode + top-k
// =====================================================================
struct EncShared {
    uint4          A4h[TILE_M][D8];          // 32 KB x-tile hi bf16
    uint4          A4l[TILE_M][D8];          // 32 KB x-tile lo bf16
    uint4          B4h[2][TILE_N][4];        // 16 KB W chunk hi [n][32k], double buffered
    uint4          B4l[2][TILE_N][4];        // 16 KB W chunk lo
    float          qv[TILE_M * TILE_N];      //  8 KB candidate queue values
    unsigned int   qk[TILE_M * TILE_N];      //  8 KB candidate keys (row<<13 | sidx)
    float          hv[TILE_M][TOP_K];        //  2 KB per-row min-heaps (values)
    int            hx[TILE_M][TOP_K];        //  2 KB per-row min-heaps (indices)
    float          thr[TILE_M];
    int            cnt[TILE_M];
    int            qcount;
};

__global__ __launch_bounds__(256) void sae_encode_topk(
    const float* __restrict__ x,
    const unsigned short* __restrict__ WtH,  // [g][s][d] bf16 hi
    const unsigned short* __restrict__ WtL,  // [g][s][d] bf16 lo
    const float* __restrict__ b_enc,
    float* __restrict__ full_acts,           // [BATCH][N_GROUPS*SUB_SIZE]
    float* __restrict__ tk_val,
    int*   __restrict__ tk_idx)
{
    __shared__ EncShared sh;
    const int tid  = threadIdx.x;
    const int lane = tid & 31;
    const int wave = tid >> 5;
    const int g    = blockIdx.x >> 7;
    const int b0   = (blockIdx.x & 127) * TILE_M;

    if (tid < TILE_M) { sh.thr[tid] = -__builtin_inff(); sh.cnt[tid] = 0; }
    if (tid == 0) sh.qcount = 0;

    // ---- preload x tile (16x1024) as packed bf16 hi/lo ----
    {
        const int r  = tid >> 4;
        const int c0 = (tid & 15) * 64;
        const float* xp = x + (size_t)(b0 + r) * D_MODEL + c0;
        for (int c8 = 0; c8 < 8; ++c8) {
            unsigned hw[4], lw[4];
            #pragma unroll
            for (int p = 0; p < 4; ++p) {
                unsigned short h0, l0, h1, l1;
                split_bf16(xp[c8*8 + 2*p    ], h0, l0);
                split_bf16(xp[c8*8 + 2*p + 1], h1, l1);
                hw[p] = (unsigned)h0 | ((unsigned)h1 << 16);
                lw[p] = (unsigned)l0 | ((unsigned)l1 << 16);
            }
            sh.A4h[r][c0/8 + c8] = make_uint4(hw[0], hw[1], hw[2], hw[3]);
            sh.A4l[r][c0/8 + c8] = make_uint4(lw[0], lw[1], lw[2], lw[3]);
        }
    }
    __syncthreads();

    const int ncol      = wave * 16;
    const int colInTile = ncol + (lane & 15);
    const int mbase     = (lane >> 4) * 8;
    const int ln  = tid >> 1;                    // B staging: row 0..127
    const int lj  = (tid & 1) * 2;               // B staging: uint4 pair 0 or 2
    const unsigned short* WgH = WtH + (size_t)g * SUB_SIZE * D_MODEL;
    const unsigned short* WgL = WtL + (size_t)g * SUB_SIZE * D_MODEL;

    for (int sc = 0; sc < NS_CHUNKS; ++sc) {
        const int s0 = sc * TILE_N;
        v8f acc1 = {0.f, 0.f, 0.f, 0.f, 0.f, 0.f, 0.f, 0.f};   // hi*hi chain
        v8f acc2 = {0.f, 0.f, 0.f, 0.f, 0.f, 0.f, 0.f, 0.f};   // hi*lo + lo*hi chain

        // stage one 128x32 bf16 chunk (hi+lo): 4 global b128 + 4 ds b128 per thread
        auto loadB = [&](int kc, int buf) {
            const size_t roff = (size_t)(s0 + ln) * D_MODEL + kc * KCHUNK;  // shorts, 64B aligned
            const uint4* s4h = (const uint4*)(WgH + roff);
            const uint4* s4l = (const uint4*)(WgL + roff);
            sh.B4h[buf][ln][lj    ] = s4h[lj    ];
            sh.B4h[buf][ln][lj + 1] = s4h[lj + 1];
            sh.B4l[buf][ln][lj    ] = s4l[lj    ];
            sh.B4l[buf][ln][lj + 1] = s4l[lj + 1];
        };

        loadB(0, 0);
        __syncthreads();

        for (int kc = 0; kc < NK_CHUNKS; ++kc) {
            const int cur = kc & 1;
            if (kc + 1 < NK_CHUNKS) loadB(kc + 1, cur ^ 1);

            BFrag ahi, alo, bhi, blo;
            const int m  = lane & 15;
            const int ka = kc * 4 + (lane >> 4);     // uint4 index of 8-short group
            ahi.q[0] = sh.A4h[m][ka];
            ahi.q[1] = sh.A4h[m][ka + 2];
            alo.q[0] = sh.A4l[m][ka];
            alo.q[1] = sh.A4l[m][ka + 2];
            const int n  = colInTile;
            const int kb = (lane >> 4);
            bhi.q[0] = sh.B4h[cur][n][kb];
            bhi.q[1] = sh.B4h[cur][n][kb + 2];
            blo.q[0] = sh.B4l[cur][n][kb];
            blo.q[1] = sh.B4l[cur][n][kb + 2];

            // bf16x3: two independent accumulator chains for ILP
            acc1 = __builtin_amdgcn_wmma_f32_16x16x32_bf16(false, ahi.v, false, bhi.v, (short)0, acc1, false, false);
            acc2 = __builtin_amdgcn_wmma_f32_16x16x32_bf16(false, ahi.v, false, blo.v, (short)0, acc2, false, false);
            acc2 = __builtin_amdgcn_wmma_f32_16x16x32_bf16(false, alo.v, false, bhi.v, (short)0, acc2, false, false);
            __syncthreads();
        }

        // ---- threshold-filtered candidate push ----
        const float bias = b_enc[g * SUB_SIZE + s0 + colInTile];
        const int   sidx = s0 + colInTile;
        #pragma unroll
        for (int i = 0; i < 8; ++i) {
            float v  = (acc1[i] + acc2[i]) + bias;
            int  row = mbase + i;
            if (v > sh.thr[row]) {
                int pos = atomicAdd(&sh.qcount, 1);
                sh.qv[pos] = v;
                sh.qk[pos] = ((unsigned)row << 13) | (unsigned)sidx;
            }
        }
        __syncthreads();

        // ---- per-row min-heap maintenance (one lane per row) ----
        if (tid < TILE_M) {
            const int r  = tid;
            const int qn = sh.qcount;
            float* hv = &sh.hv[r][0];
            int*   hx = &sh.hx[r][0];
            int c = sh.cnt[r];
            for (int j = 0; j < qn; ++j) {
                unsigned key = sh.qk[j];
                if ((int)(key >> 13) != r) continue;
                float v  = sh.qv[j];
                int   si = (int)(key & 8191u);
                if (c < TOP_K) {
                    int i2 = c++;
                    hv[i2] = v; hx[i2] = si;
                    while (i2 > 0) {
                        int p = (i2 - 1) >> 1;
                        if (hv[p] <= hv[i2]) break;
                        float tv = hv[p]; hv[p] = hv[i2]; hv[i2] = tv;
                        int   ti = hx[p]; hx[p] = hx[i2]; hx[i2] = ti;
                        i2 = p;
                    }
                    if (c == TOP_K) sh.thr[r] = hv[0];
                } else if (v > hv[0]) {
                    hv[0] = v; hx[0] = si;
                    int i2 = 0;
                    while (true) {
                        int lc = 2*i2 + 1, rc = 2*i2 + 2, m2 = i2;
                        if (lc < TOP_K && hv[lc] < hv[m2]) m2 = lc;
                        if (rc < TOP_K && hv[rc] < hv[m2]) m2 = rc;
                        if (m2 == i2) break;
                        float tv = hv[m2]; hv[m2] = hv[i2]; hv[i2] = tv;
                        int   ti = hx[m2]; hx[m2] = hx[i2]; hx[i2] = ti;
                        i2 = m2;
                    }
                    sh.thr[r] = hv[0];
                }
            }
            sh.cnt[r] = c;
        }
        if (tid == 0) sh.qcount = 0;
        __syncthreads();
    }

    // ---- emit: scatter relu'd top-k + compact lists for decode ----
    for (int e = tid; e < TILE_M * TOP_K; e += 256) {
        int r = e >> 5, slot = e & 31;
        float v  = sh.hv[r][slot];
        int   si = sh.hx[r][slot];
        float rv = v > 0.f ? v : 0.f;
        int   b  = b0 + r;
        full_acts[(size_t)b * (N_GROUPS * SUB_SIZE) + g * SUB_SIZE + si] = rv;
        size_t o = ((size_t)(g * BATCH + b)) * TOP_K + slot;
        tk_val[o] = rv;
        tk_idx[o] = si;
    }
}

__global__ void zero_kernel(float4* __restrict__ p, long long n4) {
    long long i = (long long)blockIdx.x * blockDim.x + threadIdx.x;
    long long stride = (long long)gridDim.x * blockDim.x;
    float4 z = make_float4(0.f, 0.f, 0.f, 0.f);
    for (; i < n4; i += stride) p[i] = z;
}

__global__ __launch_bounds__(256) void sae_decode(
    const float* __restrict__ W_dec,
    const float* __restrict__ tk_val,
    const int*   __restrict__ tk_idx,
    float* __restrict__ recon)
{
    const int b = blockIdx.x;
    const int d = threadIdx.x * 4;
    float4 acc = make_float4(0.f, 0.f, 0.f, 0.f);
    #pragma unroll
    for (int g = 0; g < N_GROUPS; ++g) {
        const size_t base = ((size_t)(g * BATCH + b)) * TOP_K;
        for (int i = 0; i < TOP_K; ++i) {
            float v  = tk_val[base + i];
            int   si = tk_idx[base + i];
            const float4 w = *(const float4*)(W_dec + ((size_t)g * SUB_SIZE + si) * D_MODEL + d);
            acc.x += v * w.x; acc.y += v * w.y; acc.z += v * w.z; acc.w += v * w.w;
        }
    }
    *(float4*)(recon + (size_t)b * D_MODEL + d) = acc;
}

extern "C" void kernel_launch(void* const* d_in, const int* in_sizes, int n_in,
                              void* d_out, int out_size, void* d_ws, size_t ws_size,
                              hipStream_t stream) {
    const float* x     = (const float*)d_in[0];
    const float* W_enc = (const float*)d_in[1];
    const float* b_enc = (const float*)d_in[2];
    const float* W_dec = (const float*)d_in[3];

    float* recon     = (float*)d_out;                                // [BATCH][D_MODEL]
    float* full_acts = recon + (size_t)BATCH * D_MODEL;              // [BATCH][G*SUB]

    // workspace layout: 2 MB tk_val | 2 MB tk_idx | 128 MB Wt_hi | 128 MB Wt_lo
    float*          tk_val = (float*)d_ws;
    int*            tk_idx = (int*)((char*)d_ws + (2u << 20));
    unsigned short* WtH    = (unsigned short*)((char*)d_ws + (4u << 20));
    unsigned short* WtL    = WtH + (size_t)N_GROUPS * SUB_SIZE * D_MODEL;

    // Pass 0: convert + transpose W_enc to bf16 hi/lo
    convert_wenc<<<N_GROUPS * (D_MODEL/64) * (SUB_SIZE/64), 256, 0, stream>>>(W_enc, WtH, WtL);

    // Zero the (sparse) full_acts output region
    const long long n4 = (long long)BATCH * N_GROUPS * SUB_SIZE / 4;
    zero_kernel<<<4096, 256, 0, stream>>>((float4*)full_acts, n4);

    // Fused encode GEMM (bf16x3 WMMA) + top-k
    sae_encode_topk<<<N_GROUPS * (BATCH / TILE_M), 256, 0, stream>>>(
        x, WtH, WtL, b_enc, full_acts, tk_val, tk_idx);

    // Sparse decode
    sae_decode<<<BATCH, 256, 0, stream>>>(W_dec, tk_val, tk_idx, recon);
}
